// DRRN_11476152614937
// MI455X (gfx1250) — matrix operand order repaired and checked by reading, not compiled, not run on hardware
//
#include <hip/hip_runtime.h>
#include <hip/hip_bf16.h>

// ---------------- problem constants (from reference) ----------------
#define VSZ   32000
#define EDIM  128
#define HDIM  256
#define G3H   768     // 3*H
#define BSZ   256
#define SEQ   256
#define NA    8
#define SA    32
#define NACT  (BSZ*NA)   // 2048
#define K4H   1024       // 4*H
#define MT    32         // batch rows per scan workgroup (2 WMMA M-tiles)

typedef __attribute__((ext_vector_type(16))) _Float16 v16h;
typedef __attribute__((ext_vector_type(8)))  float    v8f;

// Explicit global (addrspace 1) pointer: guarantees global_load_* (pure
// LOADcnt path) instead of flat_load_* (which also burns DScnt and the LDS
// datapath on CDNA5).
typedef const __attribute__((address_space(1))) _Float16 gf16;

// ---------------- WMMA fragment helpers (CDNA5 wave32 layouts) -------
// A 16x32 f16: lane m=l&15; lanes<16 hold K klo..klo+15,
// lanes>=16 hold K klo+16..klo+31 (matches B fragment order).
__device__ __forceinline__ v16h load_A_lds(const _Float16* base, int stride, int klo, int lane) {
    const _Float16* row = base + (lane & 15) * stride + klo + ((lane & 16) ? 16 : 0);
    v16h a;
#pragma unroll
    for (int i = 0; i < 16; ++i) a[i] = row[i];
    return a;
}

// B 32x16 f16: lane column n = n0+(l&15); lanes<16 hold K klo..klo+15,
// lanes>=16 hold K klo+16..klo+31 (16 contiguous halves per lane).
// W stored row-major [N][K] (so B[k][n] = W[n][k], i.e. computes X @ W^T).
// The 32-bit element offset is routed through an empty volatile asm so LICM
// cannot hoist the (t-invariant) loads out of the timestep loop and spill.
__device__ __forceinline__ v16h load_B_g(const _Float16* W, int ldk, int n0, int klo, int lane) {
    int off = (n0 + (lane & 15)) * ldk + klo + ((lane & 16) ? 16 : 0);
    asm volatile("" : "+v"(off));
    gf16* row = (gf16*)W + off;
    v16h b;
#pragma unroll
    for (int i = 0; i < 16; ++i) b[i] = row[i];
    return b;
}

__device__ __forceinline__ v8f wmma16(v16h a, v16h b, v8f c) {
    return __builtin_amdgcn_wmma_f32_16x16x32_f16(false, a, false, b, (short)0, c, false, false);
}

__device__ __forceinline__ float sigm(float x) { return 1.0f / (1.0f + __expf(-x)); }

// ---------------- kernel 0: convert weights f32 -> f16 ---------------
__global__ __launch_bounds__(256) void cvt_weights(const float* Wih, const float* Whh,
                                                   const float* hW,
                                                   _Float16* wih_h, _Float16* whh_h, _Float16* hw_h) {
    const int nWih = 4 * G3H * EDIM;   // 393216
    const int nWhh = 4 * G3H * HDIM;   // 786432
    const int nHW  = HDIM * K4H;       // 262144
    int i = blockIdx.x * 256 + threadIdx.x;
    if (i < nWih) { wih_h[i] = (_Float16)Wih[i]; return; }
    i -= nWih;
    if (i < nWhh) { whh_h[i] = (_Float16)Whh[i]; return; }
    i -= nWhh;
    if (i < nHW)  { hw_h[i] = (_Float16)hW[i]; }
}

// ---------------- kernel 1: fused embed + GRU scan -------------------
// One workgroup (8 waves) owns a 32-row batch tile (two WMMA M-tiles) of one
// encoder and runs the full time scan with h resident in LDS. Wave w owns
// h-column tiles {2w, 2w+1} (processed sequentially); each weight B-fragment
// feeds two WMMAs (one per M-tile) to double arithmetic intensity vs L2.
// r/z gates use merged gi+gh accumulators (4 accumulators per tile, not 6).
__global__ __launch_bounds__(256) void gru_scan(
    const int* __restrict__ obs_tok,  const int* __restrict__ obs_len,
    const int* __restrict__ look_tok, const int* __restrict__ look_len,
    const int* __restrict__ inv_tok,  const int* __restrict__ inv_len,
    const int* __restrict__ act_tok,  const int* __restrict__ act_len,
    const float* __restrict__ emb,
    const _Float16* __restrict__ wih_h, const _Float16* __restrict__ whh_h,
    const float* __restrict__ bih, const float* __restrict__ bhh,
    float* __restrict__ obs_out, float* __restrict__ look_out,
    float* __restrict__ inv_out, float* __restrict__ act_out) {

    __shared__ float    h32[MT][HDIM + 1];   // master state, f32 (~33KB)
    __shared__ _Float16 h16[MT][HDIM + 8];   // WMMA A operand (padded) (~17KB)
    __shared__ _Float16 x16[MT][EDIM + 8];   // gathered embeddings (~8.7KB)
    __shared__ int      toks[MT];
    __shared__ int      lens[MT];

    const int tid  = threadIdx.x;
    const int wave = tid >> 5;
    const int lane = tid & 31;

    int blk = blockIdx.x;
    int enc, tile, S_e;
    const int *tok, *len;
    float* outp;
    if (blk < 8)       { enc = 0; tile = blk;      tok = obs_tok;  len = obs_len;  outp = obs_out;  S_e = SEQ; }
    else if (blk < 16) { enc = 1; tile = blk - 8;  tok = look_tok; len = look_len; outp = look_out; S_e = SEQ; }
    else if (blk < 24) { enc = 2; tile = blk - 16; tok = inv_tok;  len = inv_len;  outp = inv_out;  S_e = SEQ; }
    else               { enc = 3; tile = blk - 24; tok = act_tok;  len = act_len;  outp = act_out;  S_e = SA;  }
    const int row0 = tile * MT;

    const _Float16* WihE = wih_h + (size_t)enc * G3H * EDIM;
    const _Float16* WhhE = whh_h + (size_t)enc * G3H * HDIM;
    const float* bihE = bih + enc * G3H;
    const float* bhhE = bhh + enc * G3H;

    // init state + lengths
    for (int i = tid; i < MT * HDIM; i += 256) h32[i >> 8][i & 255] = 0.0f;
    if (tid < MT) { int l = len[row0 + tid]; lens[tid] = (l < 1) ? 1 : l; }
    __syncthreads();

    for (int t = 0; t < S_e; ++t) {
        // stage tokens + refresh f16 copy of h
        if (tid < MT) toks[tid] = tok[(size_t)(row0 + tid) * S_e + t];
        for (int i = tid; i < MT * HDIM; i += 256) {
            int m = i >> 8, c = i & 255;
            h16[m][c] = (_Float16)h32[m][c];
        }
        __syncthreads();
        // gather x_t = emb[tokens]: 32 rows x 128 cols, 8 floats per thread, 2 rounds
        for (int i = tid * 8; i < MT * EDIM; i += 256 * 8) {
            int m  = i >> 7;       // /EDIM
            int c0 = i & 127;
            const float* er = emb + (size_t)toks[m] * EDIM + c0;
#pragma unroll
            for (int j = 0; j < 8; ++j) x16[m][c0 + j] = (_Float16)er[j];
        }
        __syncthreads();

#pragma unroll 1
        for (int j = 0; j < 2; ++j) {
            const int ct   = 2 * wave + j;     // h-column tile 0..15
            const int ncol = lane & 15;
            const int nr = ct * 16 + ncol;          // r-gate col
            const int nz = (16 + ct) * 16 + ncol;   // z-gate col
            const int nn = (32 + ct) * 16 + ncol;   // n-gate col

            // accumulators per M-tile: r (gi+gh merged), z (merged), inn, hn
            v8f aR[2], aZ[2], aIN[2], aHN[2];
            {
                float br = bihE[nr] + bhhE[nr];
                float bz = bihE[nz] + bhhE[nz];
                float bi = bihE[nn];
                float bh = bhhE[nn];
#pragma unroll
                for (int mt = 0; mt < 2; ++mt)
#pragma unroll
                    for (int r = 0; r < 8; ++r) {
                        aR[mt][r] = br; aZ[mt][r] = bz; aIN[mt][r] = bi; aHN[mt][r] = bh;
                    }
            }
            // gi = x @ Wih^T   (K = 128), accumulates into merged r/z + inn
#pragma unroll
            for (int k = 0; k < EDIM; k += 32) {
                v16h a0 = load_A_lds(&x16[0][0],  EDIM + 8, k, lane);
                v16h a1 = load_A_lds(&x16[16][0], EDIM + 8, k, lane);
                v16h b;
                b = load_B_g(WihE, EDIM, ct * 16, k, lane);
                aR[0] = wmma16(a0, b, aR[0]);   aR[1] = wmma16(a1, b, aR[1]);
                b = load_B_g(WihE, EDIM, (16 + ct) * 16, k, lane);
                aZ[0] = wmma16(a0, b, aZ[0]);   aZ[1] = wmma16(a1, b, aZ[1]);
                b = load_B_g(WihE, EDIM, (32 + ct) * 16, k, lane);
                aIN[0] = wmma16(a0, b, aIN[0]); aIN[1] = wmma16(a1, b, aIN[1]);
            }
            // gh = h @ Whh^T   (K = 256), accumulates into merged r/z + hn
#pragma unroll
            for (int k = 0; k < HDIM; k += 32) {
                v16h a0 = load_A_lds(&h16[0][0],  HDIM + 8, k, lane);
                v16h a1 = load_A_lds(&h16[16][0], HDIM + 8, k, lane);
                v16h b;
                b = load_B_g(WhhE, HDIM, ct * 16, k, lane);
                aR[0] = wmma16(a0, b, aR[0]);   aR[1] = wmma16(a1, b, aR[1]);
                b = load_B_g(WhhE, HDIM, (16 + ct) * 16, k, lane);
                aZ[0] = wmma16(a0, b, aZ[0]);   aZ[1] = wmma16(a1, b, aZ[1]);
                b = load_B_g(WhhE, HDIM, (32 + ct) * 16, k, lane);
                aHN[0] = wmma16(a0, b, aHN[0]); aHN[1] = wmma16(a1, b, aHN[1]);
            }
            // elementwise GRU update on the C-fragment layout
            const int mbase = (lane & 16) ? 8 : 0;
#pragma unroll
            for (int mt = 0; mt < 2; ++mt)
#pragma unroll
                for (int r = 0; r < 8; ++r) {
                    int m = mt * 16 + r + mbase;
                    int n = ct * 16 + ncol;
                    float hold = h32[m][n];
                    float rg = sigm(aR[mt][r]);
                    float zg = sigm(aZ[mt][r]);
                    float ng = tanhf(aIN[mt][r] + rg * aHN[mt][r]);
                    float hnew = (1.0f - zg) * ng + zg * hold;
                    h32[m][n] = (t < lens[m]) ? hnew : hold;
                }
        }
        __syncthreads();
    }

    // write final hidden state
    for (int i = tid; i < MT * HDIM; i += 256) {
        int m = i >> 8, c = i & 255;
        outp[(size_t)(row0 + m) * HDIM + c] = h32[m][c];
    }
}

// ---------------- kernel 2: head GEMM z = relu(concat @ W^T + b) -----
__global__ __launch_bounds__(256) void head_gemm(
    const float* __restrict__ obs_out, const float* __restrict__ look_out,
    const float* __restrict__ inv_out, const float* __restrict__ act_out,
    const _Float16* __restrict__ hw_h, const float* __restrict__ hidden_b,
    float* __restrict__ z_out) {

    __shared__ _Float16 aT[16][K4H + 8];   // 16 rows x 1024 cols, f16

    const int tid = threadIdx.x, wave = tid >> 5, lane = tid & 31;
    const int row0 = blockIdx.x * 16;

    for (int i = tid; i < 16 * K4H; i += 256) {
        int m = i >> 10, c = i & 1023;
        int row = row0 + m, b = row >> 3;   // A=8 actions per state
        float v;
        if (c < 256)       v = obs_out[b * HDIM + c];
        else if (c < 512)  v = look_out[b * HDIM + (c - 256)];
        else if (c < 768)  v = inv_out[b * HDIM + (c - 512)];
        else               v = act_out[(size_t)row * HDIM + (c - 768)];
        aT[m][c] = (_Float16)v;
    }
    __syncthreads();

#pragma unroll 1
    for (int j = 0; j < 2; ++j) {
        const int nt = wave * 2 + j;   // 0..15
        const int ncol = lane & 15;
        float bb = hidden_b[nt * 16 + ncol];
        v8f c;
#pragma unroll
        for (int r = 0; r < 8; ++r) c[r] = bb;
#pragma unroll
        for (int k = 0; k < K4H; k += 32) {
            v16h a = load_A_lds(&aT[0][0], K4H + 8, k, lane);
            c = wmma16(a, load_B_g(hw_h, K4H, nt * 16, k, lane), c);
        }
        const int mbase = (lane & 16) ? 8 : 0;
#pragma unroll
        for (int r = 0; r < 8; ++r) {
            float zv = c[r];
            zv = zv > 0.0f ? zv : 0.0f;
            z_out[(size_t)(row0 + r + mbase) * HDIM + nt * 16 + ncol] = zv;
        }
    }
}

// ---------------- kernel 3: scorer q = z @ sW^T + sb -----------------
__global__ __launch_bounds__(256) void scorer(const float* __restrict__ z,
                                              const float* __restrict__ sW,
                                              const float* __restrict__ sB,
                                              float* __restrict__ q) {
    int row  = blockIdx.x * 8 + (threadIdx.x >> 5);
    int lane = threadIdx.x & 31;
    const float* zr = z + (size_t)row * HDIM;
    float p = 0.0f;
#pragma unroll
    for (int c = lane; c < HDIM; c += 32) p += zr[c] * sW[c];
#pragma unroll
    for (int off = 16; off > 0; off >>= 1) p += __shfl_xor(p, off, 32);
    if (lane == 0) q[row] = p + sB[0];
}

// ---------------- host launcher --------------------------------------
extern "C" void kernel_launch(void* const* d_in, const int* in_sizes, int n_in,
                              void* d_out, int out_size, void* d_ws, size_t ws_size,
                              hipStream_t stream) {
    const int*   obs_tok  = (const int*)d_in[0];
    const int*   obs_len  = (const int*)d_in[1];
    const int*   look_tok = (const int*)d_in[2];
    const int*   look_len = (const int*)d_in[3];
    const int*   inv_tok  = (const int*)d_in[4];
    const int*   inv_len  = (const int*)d_in[5];
    const int*   act_tok  = (const int*)d_in[6];
    const int*   act_len  = (const int*)d_in[7];
    const float* emb      = (const float*)d_in[8];
    const float* Wih      = (const float*)d_in[9];
    const float* Whh      = (const float*)d_in[10];
    const float* bih      = (const float*)d_in[11];
    const float* bhh      = (const float*)d_in[12];
    const float* hidden_W = (const float*)d_in[13];
    const float* hidden_b = (const float*)d_in[14];
    const float* scorer_W = (const float*)d_in[15];
    const float* scorer_b = (const float*)d_in[16];
    float* q = (float*)d_out;

    // workspace layout (bytes), ~7.5 MB total
    char* ws = (char*)d_ws;
    size_t off = 0;
    _Float16* wih_h = (_Float16*)(ws + off); off += (size_t)4 * G3H * EDIM * 2;   // 786432
    _Float16* whh_h = (_Float16*)(ws + off); off += (size_t)4 * G3H * HDIM * 2;   // 1572864
    _Float16* hw_h  = (_Float16*)(ws + off); off += (size_t)HDIM * K4H * 2;       // 524288
    float* obs_out  = (float*)(ws + off);    off += (size_t)BSZ * HDIM * 4;
    float* look_out = (float*)(ws + off);    off += (size_t)BSZ * HDIM * 4;
    float* inv_out  = (float*)(ws + off);    off += (size_t)BSZ * HDIM * 4;
    float* act_out  = (float*)(ws + off);    off += (size_t)NACT * HDIM * 4;
    float* z_buf    = (float*)(ws + off);    off += (size_t)NACT * HDIM * 4;
    (void)ws_size; (void)in_sizes; (void)n_in; (void)out_size;

    // 0: weight conversion (1,441,792 elements)
    {
        int total = 4 * G3H * EDIM + 4 * G3H * HDIM + HDIM * K4H;
        int grid = (total + 255) / 256;
        cvt_weights<<<grid, 256, 0, stream>>>(Wih, Whh, hidden_W, wih_h, whh_h, hw_h);
    }
    // 1: GRU scans — 8 tiles x 3 seq encoders + 64 tiles act encoder (M=32 each)
    gru_scan<<<88, 256, 0, stream>>>(obs_tok, obs_len, look_tok, look_len,
                                     inv_tok, inv_len, act_tok, act_len,
                                     emb, wih_h, whh_h, bih, bhh,
                                     obs_out, look_out, inv_out, act_out);
    // 2: head GEMM (2048 rows -> 128 M-tiles)
    head_gemm<<<NACT / 16, 256, 0, stream>>>(obs_out, look_out, inv_out, act_out,
                                             hw_h, hidden_b, z_buf);
    // 3: scorer reduction (8 rows per block)
    scorer<<<NACT / 8, 256, 0, stream>>>(z_buf, scorer_W, scorer_b, q);
}
